// SFNN_86612310491679
// MI455X (gfx1250) — matrix operand order repaired and checked by reading, not compiled, not run on hardware
//
#include <hip/hip_runtime.h>

// ---------------------------------------------------------------------------
// SFNN forward on gfx1250: GRU edge updates via v_wmma_f32_16x16x32_bf16.
// Channel layout chosen so each lane's WMMA D fragment contains all four GRU
// gate values (r,z,xn,hn) for its 4 dims -> no cross-lane shuffles at all.
// ---------------------------------------------------------------------------

#define NN   1024      // total neurons
#define DD   8         // neuron size
#define II   64        // input layer
#define OO   64        // output layer
#define HN   (NN - II - OO)   // 896
#define TICKS 2

typedef __attribute__((ext_vector_type(16))) __bf16 v16bf;
typedef __attribute__((ext_vector_type(8)))  __bf16 v8bf;
typedef __attribute__((ext_vector_type(8)))  float  v8f;

__device__ __forceinline__ float fast_rcp(float x) {
    return __builtin_amdgcn_rcpf(x);
}
__device__ __forceinline__ float fast_sigmoid(float x) {
    return fast_rcp(1.0f + __expf(-x));
}
__device__ __forceinline__ float fast_tanh(float x) {
#if __has_builtin(__builtin_amdgcn_tanhf)
    return __builtin_amdgcn_tanhf(x);
#elif __has_builtin(__builtin_amdgcn_tanh_f32)
    return __builtin_amdgcn_tanh_f32(x);
#else
    // branch-free: tanh(x) = 1 - 2/(exp(2x)+1)
    float e = __expf(2.0f * x);
    return 1.0f - 2.0f * fast_rcp(e + 1.0f);
#endif
}

// ---------------------------------------------------------------------------
// Pack fused GRU weights W' (32 channels x 32 K) as bf16.
// Channel ch: tile = ch>>4, m = ch&15, d = (m&3) + ((m>>3)&1)*4,
//             grp = tile*2 + ((m>>2)&1)   (0=r, 1=z, 2=xn, 3=hn)
// K: [0..7]=pre, [8..15]=post, [16]=reward, [17..24]=h, [25]=bias, rest 0.
// ---------------------------------------------------------------------------
__global__ __launch_bounds__(256)
void pack_w_kernel(__bf16* __restrict__ dst,
                   const float* __restrict__ wih,  // (24,17)
                   const float* __restrict__ whh,  // (24,8)
                   const float* __restrict__ bih,  // (24)
                   const float* __restrict__ bhh)  // (24)
{
    int t = blockIdx.x * blockDim.x + threadIdx.x;
    if (t >= 32 * 32) return;
    int ch = t >> 5, k = t & 31;
    int tile = ch >> 4, m = ch & 15;
    int d   = (m & 3) + ((m >> 3) & 1) * 4;
    int grp = tile * 2 + ((m >> 2) & 1);   // 0=r, 1=z, 2=xn, 3=hn
    float v = 0.0f;
    if (k < 17) {                              // x part
        if (grp == 0)      v = wih[d * 17 + k];
        else if (grp == 1) v = wih[(8 + d) * 17 + k];
        else if (grp == 2) v = wih[(16 + d) * 17 + k];
    } else if (k < 25) {                       // h part
        int kh = k - 17;
        if (grp == 0)      v = whh[d * 8 + kh];
        else if (grp == 1) v = whh[(8 + d) * 8 + kh];
        else if (grp == 3) v = whh[(16 + d) * 8 + kh];
    } else if (k == 25) {                      // fused bias
        if (grp == 0)      v = bih[d] + bhh[d];
        else if (grp == 1) v = bih[8 + d] + bhh[8 + d];
        else if (grp == 2) v = bih[16 + d];
        else               v = bhh[16 + d];
    }
    dst[ch * 32 + k] = (__bf16)v;
}

// ---------------------------------------------------------------------------
// Input layer neurons: P[i,d] = tanh(obs[i] * sum_k(W[d,k]) + b[d])
// ---------------------------------------------------------------------------
__global__ __launch_bounds__(512)
void input_p_kernel(const float* __restrict__ obs,
                    const float* __restrict__ W,   // (8,8)
                    const float* __restrict__ b,
                    float* __restrict__ P)
{
    int t = threadIdx.x;          // 512 = 64*8
    int i = t >> 3, d = t & 7;
    float s = 0.0f;
    #pragma unroll
    for (int k = 0; k < 8; k++) s += W[d * 8 + k];
    P[i * 8 + d] = fast_tanh(obs[i] * s + b[d]);
}

// ---------------------------------------------------------------------------
// Synapse GRU update over one layer slice using WMMA.
// Tile = 16 edges (fixed adjacency row i, 16 consecutive columns j0..j0+15).
// A0 = [r0-3,z0-3 | r4-7,z4-7], A1 = [xn0-3,hn0-3 | xn4-7,hn4-7] (bf16)
// B (32x16 bf16)  : per-edge inputs [pre|post | rew,h,1]
// Lane l<16 handles dims 0-3 of edge j0+l, lane l+16 handles dims 4-7:
//   r = c0[dd], z = c0[4+dd], xn = c1[dd], hn = c1[4+dd]   (all lane-local)
// Hold != nullptr -> merge-store mode (hidden stage, ping-pong H buffers).
// ---------------------------------------------------------------------------
__global__ __launch_bounds__(256)
void syn_kernel(const float* __restrict__ P,      // (N,8) posts
                const float* __restrict__ Hrd,    // H read  (rows off+i)
                float*       __restrict__ Hwr,    // H write (rows i)
                const float* __restrict__ Hold,   // else-values (rows i) or null
                float*       __restrict__ PH,     // (N,N,8) placeholder
                const int*   __restrict__ adj,    // (N,N)
                const __bf16* __restrict__ Wp,    // fused weights 32x32
                const float* __restrict__ lrp,
                const float* __restrict__ rewp,
                int L, int off)
{
    const int lane = threadIdx.x & 31;
    const int wid  = (blockIdx.x * blockDim.x + threadIdx.x) >> 5;
    const int nw   = (gridDim.x * blockDim.x) >> 5;
    const int nTiles = L * (NN / 16);
    const float lr  = lrp[0];
    const float rew = rewp[0];

    const int  chb  = lane & 15;
    const bool hif  = (lane >> 4) != 0;    // false: dims 0-3, true: dims 4-7
    const int  ksel = hif ? 16 : 0;
    const int  base = hif ? 4 : 0;         // dim base for this lane

    // Load fused weights into WMMA A registers (once per wave).
    v16bf a0, a1;
    {
        const __bf16* w0 = Wp + chb * 32 + ksel;
        const __bf16* w1 = Wp + (16 + chb) * 32 + ksel;
        v8bf l0 = *(const v8bf*)(w0);
        v8bf h0 = *(const v8bf*)(w0 + 8);
        v8bf l1 = *(const v8bf*)(w1);
        v8bf h1 = *(const v8bf*)(w1 + 8);
        #pragma unroll
        for (int e = 0; e < 8; e++) {
            a0[e] = l0[e]; a0[8 + e] = h0[e];
            a1[e] = l1[e]; a1[8 + e] = h1[e];
        }
    }

    for (int t = wid; t < nTiles; t += nw) {
        const int i  = t >> 6;                 // NN/16 == 64 tiles per row
        const int j0 = (t & 63) << 4;
        const int jl = j0 + chb;               // this lane's edge column

        // per-edge hidden state h (row off+i, col jl)
        v8f hv = *(const v8f*)(Hrd + ((size_t)(off + i) * NN + jl) * DD);
        // post (col jl) and wave-uniform pre (row off+i)
        v8f pv   = *(const v8f*)(P + (size_t)jl * DD);
        v8f prev = *(const v8f*)(P + (size_t)(off + i) * DD);

        // Build B operand: lanes 0-15 carry K0..15 = [pre|post],
        // lanes 16-31 carry K16..31 = [rew, h0..h7, 1, 0...].
        v16bf blo, bhi;
        #pragma unroll
        for (int d = 0; d < 8; d++) {
            blo[d]     = (__bf16)prev[d];
            blo[8 + d] = (__bf16)pv[d];
            bhi[1 + d] = (__bf16)hv[d];
        }
        bhi[0] = (__bf16)rew;
        bhi[9] = (__bf16)1.0f;
        #pragma unroll
        for (int e = 10; e < 16; e++) bhi[e] = (__bf16)0.0f;
        v16bf b = hif ? bhi : blo;

        // D = W' x C : channels(M) x edges(N), K=32 (26 used, rest zero)
        v8f c0 = {};
        v8f c1 = {};
        c0 = __builtin_amdgcn_wmma_f32_16x16x32_bf16(
                 false, a0, false, b, (short)0, c0, false, false);
        c1 = __builtin_amdgcn_wmma_f32_16x16x32_bf16(
                 false, a1, false, b, (short)0, c1, false, false);

        // GRU elementwise: fully lane-local (4 dims per lane).
        // NOTE: only static vector extracts + one cndmask per value
        // (dynamic hv[base+dd] would lower to a cmp/select ladder).
        float upd[4], phv[4];
        #pragma unroll
        for (int dd = 0; dd < 4; dd++) {
            float h  = hif ? hv[4 + dd]   : hv[dd];
            float pr = hif ? prev[4 + dd] : prev[dd];
            float r  = fast_sigmoid(c0[dd]);
            float z  = fast_sigmoid(c0[4 + dd]);
            float n  = fast_tanh(c1[dd] + r * c1[4 + dd]);
            float g  = (1.0f - z) * n + z * h;
            float u  = h + g * lr;
            upd[dd] = u;
            phv[dd] = u * pr;
        }

        const int m = adj[(size_t)(off + i) * NN + jl];
        float* hw = Hwr + ((size_t)i * NN + jl) * DD + base;
        float* pw = PH  + ((size_t)i * NN + jl) * DD + base;
        if (Hold) {     // merge mode: always write Hwr (ping-pong buffer)
            const float4 hold = *(const float4*)(Hold + ((size_t)i * NN + jl) * DD + base);
            float4 o;
            o.x = m ? upd[0] : hold.x;
            o.y = m ? upd[1] : hold.y;
            o.z = m ? upd[2] : hold.z;
            o.w = m ? upd[3] : hold.w;
            *(float4*)hw = o;
            if (m) {
                float4 p = {phv[0], phv[1], phv[2], phv[3]};
                *(float4*)pw = p;
            }
        } else if (m) { // in-place mode: masked skip-store
            float4 o = {upd[0], upd[1], upd[2], upd[3]};
            float4 p = {phv[0], phv[1], phv[2], phv[3]};
            *(float4*)hw = o;
            *(float4*)pw = p;
        }
    }
}

// ---------------------------------------------------------------------------
// Column-sum of PH over all N rows + tanh matvec -> P[col]
// ---------------------------------------------------------------------------
__global__ __launch_bounds__(256)
void colsum_neuron_kernel(const float* __restrict__ PH,
                          const float* __restrict__ W,   // (8,8)
                          const float* __restrict__ b,
                          float* __restrict__ P,
                          int col_off)
{
    __shared__ float sdata[256];
    const int c = col_off + blockIdx.x;
    const int t = threadIdx.x;
    const int d = t & 7;
    const int r0 = t >> 3;                     // 0..31
    float s = 0.0f;
    for (int r = r0; r < NN; r += 32)
        s += PH[((size_t)r * NN + c) * DD + d];
    sdata[t] = s;
    __syncthreads();
    #pragma unroll
    for (int o = 128; o >= 8; o >>= 1) {
        if (t < o) sdata[t] += sdata[t + o];
        __syncthreads();
    }
    if (t < 8) {
        float acc = b[t];
        #pragma unroll
        for (int k = 0; k < 8; k++) acc += sdata[k] * W[t * 8 + k];
        P[(size_t)c * DD + t] = fast_tanh(acc);
    }
}

// ---------------------------------------------------------------------------
// argmax over P[N-O .. N, 0] (first-max semantics)
// ---------------------------------------------------------------------------
__global__ __launch_bounds__(64)
void argmax_kernel(const float* __restrict__ P, int* __restrict__ out)
{
    __shared__ float vals[64];
    __shared__ int   idxs[64];
    const int t = threadIdx.x;
    vals[t] = P[(size_t)(NN - OO + t) * DD];
    idxs[t] = t;
    __syncthreads();
    #pragma unroll
    for (int s = 32; s > 0; s >>= 1) {
        if (t < s) {
            if (vals[t + s] > vals[t] ||
                (vals[t + s] == vals[t] && idxs[t + s] < idxs[t])) {
                vals[t] = vals[t + s];
                idxs[t] = idxs[t + s];
            }
        }
        __syncthreads();
    }
    if (t == 0) out[0] = idxs[0];
}

// ---------------------------------------------------------------------------
// Host orchestration (graph-capture safe: async copies/memsets only)
// ---------------------------------------------------------------------------
extern "C" void kernel_launch(void* const* d_in, const int* in_sizes, int n_in,
                              void* d_out, int out_size, void* d_ws, size_t ws_size,
                              hipStream_t stream)
{
    const float* obs      = (const float*)d_in[0];
    const float* reward   = (const float*)d_in[1];
    const int*   adj      = (const int*)  d_in[2];
    const float* hidden0  = (const float*)d_in[3];
    const float* post0    = (const float*)d_in[4];
    const float* lr       = (const float*)d_in[5];
    const float* fc_in_w  = (const float*)d_in[6];
    const float* fc_in_b  = (const float*)d_in[7];
    const float* fc_hid_w = (const float*)d_in[8];
    const float* fc_hid_b = (const float*)d_in[9];
    const float* fc_out_w = (const float*)d_in[10];
    const float* fc_out_b = (const float*)d_in[11];
    const float* gw[3][4] = {
        {(const float*)d_in[12], (const float*)d_in[13], (const float*)d_in[14], (const float*)d_in[15]},
        {(const float*)d_in[16], (const float*)d_in[17], (const float*)d_in[18], (const float*)d_in[19]},
        {(const float*)d_in[20], (const float*)d_in[21], (const float*)d_in[22], (const float*)d_in[23]},
    };

    char* ws = (char*)d_ws;
    const size_t szH = (size_t)NN * NN * DD * sizeof(float);   // 32 MB
    float*  HA = (float*)(ws);
    float*  HB = (float*)(ws + szH);
    float*  PH = (float*)(ws + 2 * szH);
    float*  Pw = (float*)(ws + 3 * szH);                        // (N,8)
    __bf16* Wp = (__bf16*)(ws + 3 * szH + 64 * 1024);           // 3 x 32x32 bf16

    // init persistent-per-call state
    hipMemcpyAsync(HA, hidden0, szH, hipMemcpyDeviceToDevice, stream);
    hipMemcpyAsync(Pw, post0, (size_t)NN * DD * sizeof(float),
                   hipMemcpyDeviceToDevice, stream);
    hipMemsetAsync(PH, 0, szH, stream);
    for (int s = 0; s < 3; s++)
        pack_w_kernel<<<4, 256, 0, stream>>>(Wp + s * 1024,
                                             gw[s][0], gw[s][1], gw[s][2], gw[s][3]);

    float* Hc = HA;
    float* Hn = HB;
    for (int tick = 0; tick < TICKS; tick++) {
        // input layer
        input_p_kernel<<<1, 512, 0, stream>>>(obs, fc_in_w, fc_in_b, Pw);
        syn_kernel<<<512, 256, 0, stream>>>(Pw, Hc, Hc, nullptr, PH, adj,
                                            Wp, lr, reward, II, 0);
        // hidden layer (ping-pong H: read rows off+I, write rows [0,HN))
        colsum_neuron_kernel<<<HN, 256, 0, stream>>>(PH, fc_hid_w, fc_hid_b, Pw, II);
        syn_kernel<<<1024, 256, 0, stream>>>(Pw, Hc, Hn, Hc, PH, adj,
                                             Wp + 1024, lr, reward, HN, II);
        hipMemcpyAsync(Hn + (size_t)HN * NN * DD, Hc + (size_t)HN * NN * DD,
                       (size_t)(NN - HN) * NN * DD * sizeof(float),
                       hipMemcpyDeviceToDevice, stream);
        { float* tmp = Hc; Hc = Hn; Hn = tmp; }
        // output layer
        colsum_neuron_kernel<<<OO, 256, 0, stream>>>(PH, fc_out_w, fc_out_b, Pw, NN - OO);
        syn_kernel<<<512, 256, 0, stream>>>(Pw, Hc, Hc, nullptr, PH, adj,
                                            Wp + 2048, lr, reward, OO, NN - OO);
    }
    argmax_kernel<<<1, 64, 0, stream>>>(Pw, (int*)d_out);
}
